// INT8LinearWithPrequantizedWeight_59468117180900
// MI455X (gfx1250) — 8-bit matrix operand order repaired and checked by reading, compiled, not run on hardware
//
#include <hip/hip_runtime.h>
#include <stdint.h>


#define M_DIM 8192
#define K_DIM 4096
#define N_DIM 4096

typedef int v8i  __attribute__((ext_vector_type(8)));
typedef int v4i32 __attribute__((ext_vector_type(4)));

#define AS1 __attribute__((address_space(1)))
#define AS3 __attribute__((address_space(3)))

#if defined(__AMDGCN__) && __has_builtin(__builtin_amdgcn_global_load_async_to_lds_b128) && __has_builtin(__builtin_amdgcn_s_wait_asynccnt)
#define USE_ASYNC 1
#else
#define USE_ASYNC 0
#endif

__device__ __forceinline__ int clamp_q(float f) {
    int q = __float2int_rn(f);
    q = q > 127 ? 127 : q;
    q = q < -128 ? -128 : q;
    return q;
}

__global__ void __launch_bounds__(256)
quant_rows_kernel(const float* __restrict__ src, int8_t* __restrict__ q,
                  float* __restrict__ scales) {
    const int row = blockIdx.x;
    const float4* s4 = (const float4*)(src + (size_t)row * K_DIM);

    float4 v[4];
    float amax = 0.0f;
#pragma unroll
    for (int i = 0; i < 4; ++i) {
        v[i] = s4[threadIdx.x * 4 + i];
        amax = fmaxf(amax, fmaxf(fmaxf(fabsf(v[i].x), fabsf(v[i].y)),
                                 fmaxf(fabsf(v[i].z), fabsf(v[i].w))));
    }
#pragma unroll
    for (int off = 16; off > 0; off >>= 1)
        amax = fmaxf(amax, __shfl_xor(amax, off, 32));

    __shared__ float red[8];
    const int wid = threadIdx.x >> 5;
    const int lane = threadIdx.x & 31;
    if (lane == 0) red[wid] = amax;
    __syncthreads();
    amax = red[0];
#pragma unroll
    for (int w = 1; w < 8; ++w) amax = fmaxf(amax, red[w]);
    amax = fmaxf(amax, 1e-8f);

    const float scale = amax * (1.0f / 127.0f);
    const float inv   = 127.0f / amax;
    if (threadIdx.x == 0) scales[row] = scale;

    uint32_t pk[4];
#pragma unroll
    for (int i = 0; i < 4; ++i) {
        const int q0 = clamp_q(v[i].x * inv);
        const int q1 = clamp_q(v[i].y * inv);
        const int q2 = clamp_q(v[i].z * inv);
        const int q3 = clamp_q(v[i].w * inv);
        pk[i] = (uint32_t)(q0 & 255) | ((uint32_t)(q1 & 255) << 8) |
                ((uint32_t)(q2 & 255) << 16) | ((uint32_t)(q3 & 255) << 24);
    }
    uint4 o; o.x = pk[0]; o.y = pk[1]; o.z = pk[2]; o.w = pk[3];
    ((uint4*)(q + (size_t)row * K_DIM))[threadIdx.x] = o;
}

#define TK 64
#define NKSTEPS (K_DIM / TK)

__device__ __forceinline__ void stage_tile(const int8_t* __restrict__ gq,
                                           size_t grow0, int k0,
                                           int8_t* lds, int tid) {
#pragma unroll
    for (int c = 0; c < 2; ++c) {
        const int idx = tid * 2 + c;
        const int row = idx >> 2;
        const int col = (idx & 3) << 4;
        const int8_t* src = gq + (grow0 + row) * (size_t)K_DIM + k0 + col;
        int8_t* dst = lds + row * TK + col;
#if USE_ASYNC
        __builtin_amdgcn_global_load_async_to_lds_b128(
            (AS1 v4i32*)src, (AS3 v4i32*)dst, 0, 0);
#else
        *(uint4*)dst = *(const uint4*)src;
#endif
    }
}

__device__ __forceinline__ void async_wait_all() {
#if USE_ASYNC
    __builtin_amdgcn_s_wait_asynccnt(0);
#endif
}

__device__ __forceinline__ v8i load_a_frag(const int8_t* As, int mbase, int lane) {
    const int r = lane & 15, hi = lane >> 4;
    const int8_t* rowp = As + (mbase + r) * TK;
    v8i a;
#pragma unroll
    for (int j = 0; j < 8; ++j) {
        const int dw = (j & 3) + (hi << 2) + ((j >> 2) << 3);
        a[j] = *(const int*)(rowp + dw * 4);
    }
    return a;
}

__device__ __forceinline__ v8i load_b_frag(const int8_t* Bs, int nbase, int lane) {
    const int n = nbase + (lane & 15), hi = lane >> 4;
    const int8_t* colp = Bs + n * TK;
    v8i b;
#pragma unroll
    for (int j = 0; j < 8; ++j) {
        const int k = ((j & 3) << 2) + (hi << 4) + ((j >> 2) << 5);
        b[j] = *(const int*)(colp + k);
    }
    return b;
}

__global__ void __launch_bounds__(256)
gemm_i8_kernel(const int8_t* __restrict__ xq, const int8_t* __restrict__ wq,
               const float* __restrict__ xs, const float* __restrict__ wsc,
               const float* __restrict__ bias, float* __restrict__ out) {
    __shared__ __align__(16) int8_t As[2][128 * TK];
    __shared__ __align__(16) int8_t Bs[2][128 * TK];

    const int tid = threadIdx.x;
    const int lane = tid & 31;
    const int wid = tid >> 5;
    const int wave_m = wid >> 1;
    const int wave_n = wid & 1;
    const size_t m0 = (size_t)blockIdx.y * 128;
    const size_t n0 = (size_t)blockIdx.x * 128;

    v8i acc[2][4] = {};

    stage_tile(xq, m0, 0, As[0], tid);
    stage_tile(wq, n0, 0, Bs[0], tid);

#pragma clang loop unroll_count(1)
    for (int kt = 0; kt < NKSTEPS; ++kt) {
        const int cur = kt & 1;
        async_wait_all();
        __syncthreads();
        if (kt + 1 < NKSTEPS) {
            stage_tile(xq, m0, (kt + 1) * TK, As[cur ^ 1], tid);
            stage_tile(wq, n0, (kt + 1) * TK, Bs[cur ^ 1], tid);
        }

        v8i afrag[2];
#pragma unroll
        for (int mt = 0; mt < 2; ++mt)
            afrag[mt] = load_a_frag(As[cur], wave_m * 32 + mt * 16, lane);

#pragma unroll
        for (int nt = 0; nt < 4; ++nt) {
            const v8i bfrag = load_b_frag(Bs[cur], wave_n * 64 + nt * 16, lane);
#pragma unroll
            for (int mt = 0; mt < 2; ++mt) {
                acc[mt][nt] = __builtin_amdgcn_wmma_i32_16x16x64_iu8(
                    true, afrag[mt], true, bfrag, acc[mt][nt], false, false);
            }
        }
        __syncthreads();
    }

    const int nlo = lane & 15, hi = lane >> 4;
#pragma unroll
    for (int nt = 0; nt < 4; ++nt) {
        const size_t n = n0 + (size_t)(wave_n * 64 + nt * 16 + nlo);
        const float ws_n = wsc[n];
        const float b_n = bias[n];
#pragma unroll
        for (int mt = 0; mt < 2; ++mt) {
            const size_t mbase = m0 + (size_t)(wave_m * 32 + mt * 16 + hi * 8);
#pragma unroll
            for (int r = 0; r < 8; ++r) {
                const size_t m = mbase + r;
                out[m * (size_t)N_DIM + n] =
                    (float)acc[mt][nt][r] * xs[m] * ws_n + b_n;
            }
        }
    }
}

extern "C" void kernel_launch(void* const* d_in, const int* in_sizes, int n_in,
                              void* d_out, int out_size, void* d_ws, size_t ws_size,
                              hipStream_t stream) {
    const float* x    = (const float*)d_in[0];
    const float* w    = (const float*)d_in[1];
    const float* bias = (const float*)d_in[2];
    float* out = (float*)d_out;

    int8_t* xq  = (int8_t*)d_ws;
    int8_t* wq  = xq + (size_t)M_DIM * K_DIM;
    float*  xs  = (float*)(wq + (size_t)N_DIM * K_DIM);
    float*  wsc = xs + M_DIM;

    quant_rows_kernel<<<M_DIM, 256, 0, stream>>>(x, xq, xs);
    quant_rows_kernel<<<N_DIM, 256, 0, stream>>>(w, wq, wsc);

    dim3 grid(N_DIM / 128, M_DIM / 128);
    gemm_i8_kernel<<<grid, 256, 0, stream>>>(xq, wq, xs, wsc, bias, out);
}
